// NaiveCausalSelfAttention_23029614641223
// MI455X (gfx1250) — matrix operand order repaired and checked by reading, not compiled, not run on hardware
//
#include <hip/hip_runtime.h>
#include <math.h>

typedef __attribute__((ext_vector_type(16))) __bf16 v16bf;
typedef __attribute__((ext_vector_type(8)))  float  v8f;
typedef __attribute__((ext_vector_type(4)))  unsigned uint4v;
typedef __attribute__((__vector_size__(4 * sizeof(int)))) int v4i;  // matches builtin param

constexpr int CB   = 2;     // batch
constexpr int CS   = 2048;  // sequence
constexpr int CH   = 2048;  // hidden
constexpr int CNH  = 16;    // query heads
constexpr int CNKV = 4;     // kv heads
constexpr int CHD  = 128;   // head dim

static __device__ __forceinline__ unsigned short f2bf(float f) {
    union { float f; unsigned u; } x; x.f = f;
    unsigned r = x.u + 0x7FFFu + ((x.u >> 16) & 1u);   // round-to-nearest-even
    return (unsigned short)(r >> 16);
}
static __device__ __forceinline__ float bf2f(unsigned short b) {
    union { unsigned u; float f; } x; x.u = ((unsigned)b) << 16;
    return x.f;
}

union Frag {
    v16bf  v;
    unsigned u[8];
    uint4v q[2];
};

// ---------------- CDNA5 async global->LDS copy (ASYNCcnt path), with fallback
#if defined(__has_builtin)
#  if __has_builtin(__builtin_amdgcn_global_load_async_to_lds_b128)
#    define HAVE_ASYNC_LDS 1
#  endif
#endif
#ifndef HAVE_ASYNC_LDS
#  define HAVE_ASYNC_LDS 0
#endif

static __device__ __forceinline__ void copy_b128_to_lds(const uint4v* __restrict__ g,
                                                        uint4v* l) {
#if HAVE_ASYNC_LDS
    __builtin_amdgcn_global_load_async_to_lds_b128(
        (__attribute__((address_space(1))) v4i*)g,
        (__attribute__((address_space(3))) v4i*)l,
        0, 0);
#else
    *l = *g;
#endif
}
static __device__ __forceinline__ void copy_wait() {
#if HAVE_ASYNC_LDS
#  if __has_builtin(__builtin_amdgcn_s_wait_asynccnt)
    __builtin_amdgcn_s_wait_asynccnt(0);
#  else
    asm volatile("s_wait_asynccnt 0" ::: "memory");
#  endif
#endif
}

// ---------------------------------------------------------------- convert ---
__global__ void cvt_f32_to_bf16(const float* __restrict__ src,
                                unsigned short* __restrict__ dst, int n) {
    int i = blockIdx.x * blockDim.x + threadIdx.x;
    if (i < n) dst[i] = f2bf(src[i]);
}

// ------------------------------------------------------------------- GEMM ---
// C[M,N] = A[M,K]*B[K,N]; bf16 in, fp32 accum. Block 256 thr = 8 waves,
// tile 64(M) x 256(N); wave tile 32x64 (2x4 accumulators); k-step 32.
// LDS layouts are fragment-contiguous: two ds_load_b128 per WMMA operand.
template <bool OUT_F32>
__global__ __launch_bounds__(256) void gemm_bf16_wmma(
    const unsigned short* __restrict__ A,
    const unsigned short* __restrict__ Bw,
    void* __restrict__ Cp,
    int M, int N, int K)
{
    __shared__ __align__(16) unsigned As[64 * 16];   // [row][k-pair 0..15]   4 KB
    __shared__ __align__(16) unsigned Bs[256 * 16];  // [col][k-pair 0..15]  16 KB

    const int tid  = threadIdx.x;
    const int lane = tid & 31;
    const int wave = tid >> 5;
    const int wm   = wave >> 2;          // 0..1 : 32 rows each
    const int wn   = wave & 3;           // 0..3 : 64 cols each
    const int half = lane >> 4;
    const int l16  = lane & 15;
    const int m0   = blockIdx.y * 64;
    const int n0   = blockIdx.x * 256;

    v8f zero = {0.f,0.f,0.f,0.f,0.f,0.f,0.f,0.f};
    v8f acc[2][4];
    for (int i = 0; i < 2; ++i)
        for (int j = 0; j < 4; ++j) acc[i][j] = zero;

    const uint4v* Au4 = (const uint4v*)A;
    const uint4v* Bw4 = (const uint4v*)Bw;
    const int K8 = K >> 3, N8 = N >> 3;

    for (int kt = 0; kt < K; kt += 32) {
        // --- stage A tile 64x32 bf16: one async b128 per thread
        {
            int row = tid >> 2, c4 = tid & 3;
            copy_b128_to_lds(&Au4[(size_t)(m0 + row) * K8 + (kt >> 3) + c4],
                             (uint4v*)&As[row * 16 + c4 * 4]);
        }
        // --- stage B tile 32x256 bf16 packed as [col][k-pair]
        for (int uu = tid; uu < 512; uu += 256) {
            int pp = uu >> 5, nblk = uu & 31;               // k-pair, 8-col block
            uint4v a = Bw4[(size_t)(kt + 2 * pp)     * N8 + (n0 >> 3) + nblk];
            uint4v b = Bw4[(size_t)(kt + 2 * pp + 1) * N8 + (n0 >> 3) + nblk];
#pragma unroll
            for (int j = 0; j < 8; ++j) {
                unsigned lo = (a[j >> 1] >> ((j & 1) * 16)) & 0xFFFFu;
                unsigned hi = (b[j >> 1] >> ((j & 1) * 16)) & 0xFFFFu;
                Bs[(nblk * 8 + j) * 16 + pp] = lo | (hi << 16);
            }
        }
        copy_wait();
        __syncthreads();

        Frag af[2], bfr[4];
#pragma unroll
        for (int im = 0; im < 2; ++im) {
            int row = wm * 32 + im * 16 + l16;
            af[im].q[0] = *(const uint4v*)&As[row * 16 + half * 4];
            af[im].q[1] = *(const uint4v*)&As[row * 16 + 8 + half * 4];
        }
#pragma unroll
        for (int in = 0; in < 4; ++in) {
            int n = wn * 64 + in * 16 + l16;
            bfr[in].q[0] = *(const uint4v*)&Bs[n * 16 + half * 4];
            bfr[in].q[1] = *(const uint4v*)&Bs[n * 16 + 8 + half * 4];
        }
#pragma unroll
        for (int im = 0; im < 2; ++im)
#pragma unroll
            for (int in = 0; in < 4; ++in)
                acc[im][in] = __builtin_amdgcn_wmma_f32_16x16x32_bf16(
                    false, af[im].v, false, bfr[in].v,
                    (short)0, acc[im][in], false, false);
        __syncthreads();
    }

    // C layout: VGPR v, lane -> row = v + 8*half, col = l16
    for (int im = 0; im < 2; ++im)
        for (int in = 0; in < 4; ++in)
#pragma unroll
            for (int v = 0; v < 8; ++v) {
                int row = m0 + wm * 32 + im * 16 + v + 8 * half;
                int col = n0 + wn * 64 + in * 16 + l16;
                float val = acc[im][in][v];
                if (OUT_F32) ((float*)Cp)[(size_t)row * N + col] = val;
                else ((unsigned short*)Cp)[(size_t)row * N + col] = f2bf(val);
            }
}

// ------------------------------------------------------------------- RoPE ---
__global__ void rope_scale(unsigned short* __restrict__ T,
                           const int* __restrict__ pos,
                           int nheads, float scale) {
    int idx = blockIdx.x * blockDim.x + threadIdx.x;
    int total = CB * CS * nheads * (CHD / 2);
    if (idx >= total) return;
    int d   = idx & (CHD / 2 - 1);
    int h   = (idx >> 6) % nheads;
    int row = idx / ((CHD / 2) * nheads);
    int s   = row & (CS - 1);
    float p   = (float)pos[s];
    float inv = __powf(10000.0f, -(float)(2 * d) * (1.0f / (float)CHD));
    float sn, cs;
    __sincosf(p * inv, &sn, &cs);
    size_t base = (size_t)row * (nheads * CHD) + h * CHD;
    float x1 = bf2f(T[base + d]);
    float x2 = bf2f(T[base + d + CHD / 2]);
    T[base + d]           = f2bf((x1 * cs - x2 * sn) * scale);
    T[base + d + CHD / 2] = f2bf((x2 * cs + x1 * sn) * scale);
}

// -------------------------------------------------------- flash attention ---
// Block = 128 threads (4 waves); each wave owns 16 query rows; 32-key chunks,
// online softmax, all matmuls on v_wmma_f32_16x16x32_bf16.
__global__ __launch_bounds__(128) void flash_attn_wmma(
    const unsigned short* __restrict__ Qb,   // [CB*CS][CNH*CHD]  roped, pre-scaled
    const unsigned short* __restrict__ Kb,   // [CB*CS][CNKV*CHD] roped
    const unsigned short* __restrict__ Vb,   // [CB*CS][CNKV*CHD]
    unsigned short* __restrict__ Ob)         // [CB*CS][CNH*CHD]
{
    __shared__ __align__(16) unsigned Kp[32 * 64];          // [key][hd-pair]  8 KB
    __shared__ __align__(16) unsigned Vp[128 * 16];         // [hd][key-pair]  8 KB
    __shared__ __align__(16) unsigned short Ps[4][16 * 32]; // per-wave P      4 KB

    const int blk = blockIdx.x;
    const int qt  = blk & (CS / 64 - 1);
    const int h   = (blk / (CS / 64)) & (CNH - 1);
    const int b   = blk / ((CS / 64) * CNH);
    const int kv  = h / (CNH / CNKV);
    const int tid = threadIdx.x, lane = tid & 31, wave = tid >> 5;
    const int half = lane >> 4, l16 = lane & 15;
    const int q0 = qt * 64;
    const int qr = q0 + wave * 16;

    // Preload Q fragments (A layout) with b128 global loads, kept in registers
    Frag qfr[4];
    {
        const uint4v* Qu4 = (const uint4v*)Qb;
        size_t rb4 = ((size_t)(b * CS + qr + l16) * (CNH * CHD) + h * CHD) >> 3;
#pragma unroll
        for (int hs = 0; hs < 4; ++hs) {
            qfr[hs].q[0] = Qu4[rb4 + hs * 4 + half];
            qfr[hs].q[1] = Qu4[rb4 + hs * 4 + 2 + half];
        }
    }

    v8f zero = {0.f,0.f,0.f,0.f,0.f,0.f,0.f,0.f};
    v8f o[8];
    for (int t = 0; t < 8; ++t) o[t] = zero;
    float mrow[8], lrow[8];
    for (int v = 0; v < 8; ++v) { mrow[v] = -1e30f; lrow[v] = 0.f; }

    const uint4v* Ku4 = (const uint4v*)Kb;
    const uint4v* Vu4 = (const uint4v*)Vb;
    const int nchunk = (q0 + 64) / 32;
    for (int ch = 0; ch < nchunk; ++ch) {
        const int j0 = ch * 32;
        // K chunk: direct async b128 copies, layout [key][hd-pair] (frag-contiguous)
        for (int i = tid; i < 512; i += 128) {
            int key = i >> 4, p4 = i & 15;
            copy_b128_to_lds(&Ku4[(size_t)(b * CS + j0 + key) * 64 + kv * 16 + p4],
                             (uint4v*)&Kp[key * 64 + p4 * 4]);
        }
        // V chunk: pack key-pairs, layout [hd][key-pair] (frag-contiguous)
        for (int uu = tid; uu < 256; uu += 128) {
            int pp = uu >> 4, nblk = uu & 15;
            uint4v a = Vu4[(size_t)(b * CS + j0 + 2 * pp)     * 64 + kv * 16 + nblk];
            uint4v c = Vu4[(size_t)(b * CS + j0 + 2 * pp + 1) * 64 + kv * 16 + nblk];
#pragma unroll
            for (int j = 0; j < 8; ++j) {
                unsigned lo = (a[j >> 1] >> ((j & 1) * 16)) & 0xFFFFu;
                unsigned hi = (c[j >> 1] >> ((j & 1) * 16)) & 0xFFFFu;
                Vp[(nblk * 8 + j) * 16 + pp] = lo | (hi << 16);
            }
        }
        copy_wait();
        __syncthreads();

        if (j0 <= qr + 15) {   // wave-uniform: EXEC all-ones around WMMA
            v8f sc[2]; sc[0] = zero; sc[1] = zero;
#pragma unroll
            for (int hs = 0; hs < 4; ++hs) {
#pragma unroll
                for (int t = 0; t < 2; ++t) {
                    Frag bv;
                    int key = t * 16 + l16;
                    bv.q[0] = *(const uint4v*)&Kp[key * 64 + hs * 16 + half * 4];
                    bv.q[1] = *(const uint4v*)&Kp[key * 64 + hs * 16 + 8 + half * 4];
                    sc[t] = __builtin_amdgcn_wmma_f32_16x16x32_bf16(
                        false, qfr[hs].v, false, bv.v, (short)0, sc[t], false, false);
                }
            }
            // online softmax (rows live across 16-lane groups)
#pragma unroll
            for (int v = 0; v < 8; ++v) {
                int row = qr + v + 8 * half;
                for (int t = 0; t < 2; ++t) {
                    int key = j0 + t * 16 + l16;
                    if (key > row) sc[t][v] = -1e30f;
                }
                float mx = fmaxf(sc[0][v], sc[1][v]);
                for (int off = 1; off < 16; off <<= 1)
                    mx = fmaxf(mx, __shfl_xor(mx, off, 32));
                float mn   = fmaxf(mrow[v], mx);
                float corr = __expf(mrow[v] - mn);
                float ls = 0.f;
                for (int t = 0; t < 2; ++t) {
                    float pv = __expf(sc[t][v] - mn);
                    sc[t][v] = pv;
                    ls += pv;
                }
                for (int off = 1; off < 16; off <<= 1)
                    ls += __shfl_xor(ls, off, 32);
                lrow[v] = lrow[v] * corr + ls;
                mrow[v] = mn;
                for (int t = 0; t < 8; ++t) o[t][v] *= corr;
                for (int t = 0; t < 2; ++t)
                    Ps[wave][(v + 8 * half) * 32 + t * 16 + l16] = f2bf(sc[t][v]);
            }
            // reload P as A fragment (two b128 LDS loads)
            Frag pf;
            const unsigned* Psu = (const unsigned*)Ps[wave];
            pf.q[0] = *(const uint4v*)&Psu[l16 * 16 + half * 4];
            pf.q[1] = *(const uint4v*)&Psu[l16 * 16 + 8 + half * 4];
            // O(16x128) += P(16x32) * V(32x128)
#pragma unroll
            for (int t = 0; t < 8; ++t) {
                Frag vv;
                int n = t * 16 + l16;
                vv.q[0] = *(const uint4v*)&Vp[n * 16 + half * 4];
                vv.q[1] = *(const uint4v*)&Vp[n * 16 + 8 + half * 4];
                o[t] = __builtin_amdgcn_wmma_f32_16x16x32_bf16(
                    false, pf.v, false, vv.v, (short)0, o[t], false, false);
            }
        }
        __syncthreads();
    }

    for (int t = 0; t < 8; ++t)
#pragma unroll
        for (int v = 0; v < 8; ++v) {
            int row = qr + v + 8 * half;
            int col = t * 16 + l16;
            Ob[(size_t)(b * CS + row) * (CNH * CHD) + h * CHD + col] =
                f2bf(o[t][v] / lrow[v]);
        }
}

// ----------------------------------------------------------------- launch ---
extern "C" void kernel_launch(void* const* d_in, const int* in_sizes, int n_in,
                              void* d_out, int out_size, void* d_ws, size_t ws_size,
                              hipStream_t stream) {
    (void)in_sizes; (void)n_in; (void)out_size; (void)ws_size;
    const float* x  = (const float*)d_in[0];
    const int*  pos = (const int*)d_in[1];
    const float* Wq = (const float*)d_in[2];
    const float* Wk = (const float*)d_in[3];
    const float* Wv = (const float*)d_in[4];
    const float* Wo = (const float*)d_in[5];
    float* out = (float*)d_out;

    char* ws = (char*)d_ws;
    const size_t MB = 1u << 20;
    unsigned short* xb  = (unsigned short*)(ws +  0 * MB); // 16 MB
    unsigned short* Wqb = (unsigned short*)(ws + 16 * MB); //  8 MB
    unsigned short* Wkb = (unsigned short*)(ws + 24 * MB); //  2 MB
    unsigned short* Wvb = (unsigned short*)(ws + 26 * MB); //  2 MB
    unsigned short* Wob = (unsigned short*)(ws + 28 * MB); //  8 MB
    unsigned short* Qb  = (unsigned short*)(ws + 36 * MB); // 16 MB
    unsigned short* Kb  = (unsigned short*)(ws + 52 * MB); //  4 MB
    unsigned short* Vb  = (unsigned short*)(ws + 56 * MB); //  4 MB
    unsigned short* Ob  = (unsigned short*)(ws + 60 * MB); // 16 MB

    {
        int n;
        n = CB * CS * CH;      cvt_f32_to_bf16<<<(n + 255) / 256, 256, 0, stream>>>(x,  xb,  n);
        n = CH * CNH * CHD;    cvt_f32_to_bf16<<<(n + 255) / 256, 256, 0, stream>>>(Wq, Wqb, n);
        n = CH * CNKV * CHD;   cvt_f32_to_bf16<<<(n + 255) / 256, 256, 0, stream>>>(Wk, Wkb, n);
        n = CH * CNKV * CHD;   cvt_f32_to_bf16<<<(n + 255) / 256, 256, 0, stream>>>(Wv, Wvb, n);
        n = CNH * CHD * CH;    cvt_f32_to_bf16<<<(n + 255) / 256, 256, 0, stream>>>(Wo, Wob, n);
    }

    // QKV projections (N-tile 256)
    gemm_bf16_wmma<false><<<dim3((CNH  * CHD) / 256, (CB * CS) / 64), 256, 0, stream>>>(
        xb, Wqb, Qb, CB * CS, CNH * CHD, CH);
    gemm_bf16_wmma<false><<<dim3((CNKV * CHD) / 256, (CB * CS) / 64), 256, 0, stream>>>(
        xb, Wkb, Kb, CB * CS, CNKV * CHD, CH);
    gemm_bf16_wmma<false><<<dim3((CNKV * CHD) / 256, (CB * CS) / 64), 256, 0, stream>>>(
        xb, Wvb, Vb, CB * CS, CNKV * CHD, CH);

    // RoPE (fold 1/sqrt(HD) into Q)
    {
        int nq = CB * CS * CNH  * (CHD / 2);
        int nk = CB * CS * CNKV * (CHD / 2);
        rope_scale<<<(nq + 255) / 256, 256, 0, stream>>>(Qb, pos, CNH,  0.08838834764831845f);
        rope_scale<<<(nk + 255) / 256, 256, 0, stream>>>(Kb, pos, CNKV, 1.0f);
    }

    // causal GQA flash attention
    flash_attn_wmma<<<CB * CNH * (CS / 64), 128, 0, stream>>>(Qb, Kb, Vb, Ob);

    // output projection -> fp32
    gemm_bf16_wmma<true><<<dim3(CH / 256, (CB * CS) / 64), 256, 0, stream>>>(
        Ob, Wob, out, CB * CS, CH, CNH * CHD);
}
